// PairContrastiveLoss_69174743269986
// MI455X (gfx1250) — matrix operand order repaired and checked by reading, not compiled, not run on hardware
//
#include <hip/hip_runtime.h>
#include <math.h>

typedef unsigned short u16;
typedef unsigned int   u32;
typedef unsigned long long u64;

typedef __attribute__((ext_vector_type(16))) __bf16 v16bf;
typedef __attribute__((ext_vector_type(8)))  float  v8f;
typedef __attribute__((ext_vector_type(8)))  u16    u16x8;
typedef __attribute__((ext_vector_type(4)))  float  f32x4;

// Problem constants
constexpr int Bn   = 64;            // batch
constexpr int HW   = 384;
constexpr int PB   = 16;            // patch
constexpr int NP   = 24;            // patches per spatial dim
constexpr int NPP  = NP * NP;       // 576 patches per image
constexpr int D    = 768;           // hidden (= conv out channels = K of im2col GEMM)
constexpr int L    = 1024;          // latents
constexpr int ROWS = Bn * NPP;      // 36864 patch rows
constexpr float TAU_INV = 2.0f;     // 1/0.5
constexpr float EPSV    = 1e-8f;

__device__ __forceinline__ u16 f2bf(float f) {
    u32 u = __float_as_uint(f);
    u32 r = u + 0x7FFFu + ((u >> 16) & 1u);   // round-to-nearest-even
    return (u16)(r >> 16);
}
__device__ __forceinline__ float bf2f(u16 h) {
    return __uint_as_float(((u32)h) << 16);
}

union AB16 { u16x8 h[2]; v16bf v; };

// ---- CDNA5 async global->LDS copy (ASYNCcnt path) -------------------------
__device__ __forceinline__ void async_copy_b128(u32 lds_byte_off, const void* gptr) {
    u64 ga = (u64)(uintptr_t)gptr;
    asm volatile("global_load_async_to_lds_b128 %0, %1, off"
                 :: "v"(lds_byte_off), "v"(ga) : "memory");
}
__device__ __forceinline__ void wait_async0() {
#if __has_builtin(__builtin_amdgcn_s_wait_asynccnt)
    __builtin_amdgcn_s_wait_asynccnt(0);
#else
    asm volatile("s_wait_asynccnt 0x0" ::: "memory");
#endif
}

// Stage one contiguous 24KB (16 rows x 768 bf16) A tile into LDS, 256 threads.
__device__ __forceinline__ void stage_atile(u16* atile, const u16* gsrc, int tid) {
    u32 lds_base = (u32)(uintptr_t)atile;   // low 32 bits = LDS byte offset
    #pragma unroll
    for (int i = 0; i < 6; ++i) {
        u32 off = (u32)(tid + i * 256) * 16u;   // 1536 x 16B chunks
        async_copy_b128(lds_base + off, (const char*)gsrc + off);
    }
    wait_async0();          // per-wave: own async loads done
    __syncthreads();        // cross-wave visibility
}

// ---------------------------------------------------------------- prep kernels
__global__ void cvt_bf16(const float* __restrict__ src, u16* __restrict__ dst, int n) {
    int i = blockIdx.x * blockDim.x + threadIdx.x;
    if (i < n) dst[i] = f2bf(src[i]);
}

__global__ void zerof(float* __restrict__ p, int n) {
    int i = blockIdx.x * blockDim.x + threadIdx.x;
    if (i < n) p[i] = 0.0f;
}

// im2col + fp32->bf16: each thread emits 8 contiguous k (half of one patch row)
__global__ void im2col_bf16(const float* __restrict__ x, u16* __restrict__ abf) {
    int t = blockIdx.x * blockDim.x + threadIdx.x;   // over ROWS * 96
    if (t >= ROWS * (D / 8)) return;
    int m  = t / (D / 8);
    int kc = (t % (D / 8)) * 8;
    int b = m / NPP, pidx = m % NPP;
    int py = pidx / NP, px = pidx % NP;
    int c = kc >> 8, rem = kc & 255, r = rem >> 4, col = rem & 15;
    const float* src = x + ((size_t)(b * 3 + c) * HW + py * PB + r) * HW + px * PB + col;
    f32x4 f0 = *(const f32x4*)src;
    f32x4 f1 = *(const f32x4*)(src + 4);
    u16x8 hh;
    #pragma unroll
    for (int j = 0; j < 4; ++j) { hh[j] = f2bf(f0[j]); hh[4 + j] = f2bf(f1[j]); }
    *(u16x8*)(abf + (size_t)m * D + kc) = hh;
}

// One wave32 per row: L2 norm of a bf16 row
__global__ void rownorm(const u16* __restrict__ src, float* __restrict__ out,
                        int rows, int cols) {
    int lane = threadIdx.x & 31;
    int row  = (blockIdx.x * blockDim.x + threadIdx.x) >> 5;
    if (row >= rows) return;
    const u16* r = src + (size_t)row * cols;
    float s = 0.0f;
    for (int c = lane; c < cols; c += 32) {
        float v = bf2f(r[c]);
        s += v * v;
    }
    #pragma unroll
    for (int off = 16; off >= 1; off >>= 1) s += __shfl_xor(s, off, 32);
    if (lane == 0) out[row] = sqrtf(s);
}

// ---------------------------------------------------------------- patch-embed GEMM
// P = A(bf16, im2col) x W^T + b. Block: 8 waves share one 16-row A tile (LDS-staged);
// each wave computes 6 O-tiles (16x96) -> 6 WMMAs per A fragment.
__global__ void gemm_patch(const u16* __restrict__ abf, const u16* __restrict__ wbf,
                           const float* __restrict__ bias, u16* __restrict__ pbf) {
    __shared__ u16 atile[16 * D];
    const int lane  = threadIdx.x & 31;
    const int wv    = threadIdx.x >> 5;
    const int tileM = blockIdx.x;             // 0..2303
    const int half  = lane >> 4;
    const int nc    = lane & 15;

    stage_atile(atile, abf + (size_t)tileM * 16 * D, threadIdx.x);

    v8f acc[6];
    #pragma unroll
    for (int i = 0; i < 6; ++i) acc[i] = (v8f){};

    for (int kb = 0; kb < D; kb += 32) {
        AB16 af;
        const u16* as = &atile[nc * D + kb + half * 8];
        af.h[0] = *(const u16x8*)(as);
        af.h[1] = *(const u16x8*)(as + 16);
        #pragma unroll
        for (int i = 0; i < 6; ++i) {
            int o = (wv * 6 + i) * 16 + nc;
            const u16* bs = wbf + (size_t)o * D + kb + half * 8;
            AB16 bf;
            bf.h[0] = *(const u16x8*)(bs);
            bf.h[1] = *(const u16x8*)(bs + 16);
            acc[i] = __builtin_amdgcn_wmma_f32_16x16x32_bf16(
                false, af.v, false, bf.v, (short)0, acc[i], false, false);
        }
    }
    // Epilogue: VGPR j -> M = j + half*8; N = lane%16
    #pragma unroll
    for (int i = 0; i < 6; ++i) {
        int o = (wv * 6 + i) * 16 + nc;
        float bval = bias[o];
        #pragma unroll
        for (int j = 0; j < 8; ++j) {
            int row = tileM * 16 + j + half * 8;
            pbf[(size_t)row * D + o] = f2bf(acc[i][j] + bval);
        }
    }
}

// ---------------------------------------------------------------- fused score GEMM
// dots = P(bf16) x latent^T, cosine-normalized, exp(score/tau) summed per image.
// Block: 8 waves share one LDS-staged A tile; each wave computes 4 N-tiles.
__global__ void gemm_scores(const u16* __restrict__ pbf, const u16* __restrict__ lat,
                            const float* __restrict__ pnorm, const float* __restrict__ lnorm,
                            float* __restrict__ sums) {
    __shared__ u16 atile[16 * D];
    const int lane  = threadIdx.x & 31;
    const int wv    = threadIdx.x >> 5;
    const int tileM = blockIdx.x;                    // 0..2303
    const int ntB   = (blockIdx.y * 8 + wv) * 4;     // N-tile base, 0..60
    const int half  = lane >> 4;
    const int nc    = lane & 15;

    // prefetch this wave's latent B rows (global_prefetch_b8)
    #pragma unroll
    for (int i = 0; i < 4; ++i)
        __builtin_prefetch(lat + (size_t)((ntB + i) * 16 + nc) * D, 0, 3);

    stage_atile(atile, pbf + (size_t)tileM * 16 * D, threadIdx.x);

    v8f acc[4];
    #pragma unroll
    for (int i = 0; i < 4; ++i) acc[i] = (v8f){};

    for (int kb = 0; kb < D; kb += 32) {
        AB16 af;
        const u16* as = &atile[nc * D + kb + half * 8];
        af.h[0] = *(const u16x8*)(as);
        af.h[1] = *(const u16x8*)(as + 16);
        #pragma unroll
        for (int i = 0; i < 4; ++i) {
            int n = (ntB + i) * 16 + nc;
            const u16* bs = lat + (size_t)n * D + kb + half * 8;
            AB16 bf;
            bf.h[0] = *(const u16x8*)(bs);
            bf.h[1] = *(const u16x8*)(bs + 16);
            acc[i] = __builtin_amdgcn_wmma_f32_16x16x32_bf16(
                false, af.v, false, bf.v, (short)0, acc[i], false, false);
        }
    }

    float pn[8];
    #pragma unroll
    for (int j = 0; j < 8; ++j) pn[j] = pnorm[tileM * 16 + j + half * 8];

    float s = 0.0f;
    #pragma unroll
    for (int i = 0; i < 4; ++i) {
        float ln = lnorm[(ntB + i) * 16 + nc];
        #pragma unroll
        for (int j = 0; j < 8; ++j) {
            float denom = fmaxf(pn[j] * ln, EPSV);
            s += __expf((acc[i][j] / denom) * TAU_INV);
        }
    }
    #pragma unroll
    for (int off = 16; off >= 1; off >>= 1) s += __shfl_xor(s, off, 32);
    if (lane == 0) {
        // all 16 rows of this M-tile belong to one image (576 % 16 == 0)
        atomicAdd(&sums[(tileM * 16) / NPP], s);
    }
}

// ---------------------------------------------------------------- finalize
__global__ void finalize(const float* __restrict__ sums_close,
                         const float* __restrict__ sums_far,
                         float* __restrict__ out) {
    __shared__ float red[64];
    int t = threadIdx.x;
    red[t] = __logf(sums_far[t]) - __logf(sums_close[t]);   // -log(close/far)
    __syncthreads();
    for (int s = 32; s > 0; s >>= 1) {
        if (t < s) red[t] += red[t + s];
        __syncthreads();
    }
    if (t == 0) out[0] = red[0] / (float)Bn;
}

// ---------------------------------------------------------------- launch
extern "C" void kernel_launch(void* const* d_in, const int* in_sizes, int n_in,
                              void* d_out, int out_size, void* d_ws, size_t ws_size,
                              hipStream_t stream) {
    const float* x      = (const float*)d_in[0];
    const float* conv_w = (const float*)d_in[1];   // (768, 3*16*16) row-major = (O, K)
    const float* conv_b = (const float*)d_in[2];
    const float* lat1   = (const float*)d_in[3];   // far
    const float* lat2   = (const float*)d_in[4];   // close
    float* out = (float*)d_out;

    char*  ws  = (char*)d_ws;
    size_t off = 0;
    auto alloc = [&](size_t bytes) -> char* {
        char* p = ws + off;
        off += (bytes + 255) & ~(size_t)255;
        return p;
    };
    u16*   wbf   = (u16*)alloc((size_t)D * D * 2);
    u16*   l1bf  = (u16*)alloc((size_t)L * D * 2);
    u16*   l2bf  = (u16*)alloc((size_t)L * D * 2);
    u16*   abf   = (u16*)alloc((size_t)ROWS * D * 2);   // im2col bf16
    u16*   pbf   = (u16*)alloc((size_t)ROWS * D * 2);   // patch embeddings bf16
    float* pnorm = (float*)alloc((size_t)ROWS * 4);
    float* l1n   = (float*)alloc((size_t)L * 4);
    float* l2n   = (float*)alloc((size_t)L * 4);
    float* sums  = (float*)alloc(128 * 4);              // [0..63] close, [64..127] far

    const int nW = D * D, nL = L * D, nA = ROWS * (D / 8);
    cvt_bf16<<<(nW + 255) / 256, 256, 0, stream>>>(conv_w, wbf, nW);
    cvt_bf16<<<(nL + 255) / 256, 256, 0, stream>>>(lat1, l1bf, nL);
    cvt_bf16<<<(nL + 255) / 256, 256, 0, stream>>>(lat2, l2bf, nL);
    im2col_bf16<<<(nA + 255) / 256, 256, 0, stream>>>(x, abf);
    zerof<<<1, 128, 0, stream>>>(sums, 128);

    // 2304 M-tiles; block = 8 waves x 6 O-tiles = all 48 O-tiles
    gemm_patch<<<dim3(ROWS / 16), 256, 0, stream>>>(abf, wbf, conv_b, pbf);

    rownorm<<<(ROWS * 32 + 255) / 256, 256, 0, stream>>>(pbf, pnorm, ROWS, D);
    rownorm<<<(L * 32 + 255) / 256, 256, 0, stream>>>(l1bf, l1n, L, D);
    rownorm<<<(L * 32 + 255) / 256, 256, 0, stream>>>(l2bf, l2n, L, D);

    // 2304 M-tiles x 2 groups; block = 8 waves x 4 N-tiles = 32 N-tiles per group
    gemm_scores<<<dim3(ROWS / 16, 2), 256, 0, stream>>>(pbf, l2bf, pnorm, l2n, sums);
    gemm_scores<<<dim3(ROWS / 16, 2), 256, 0, stream>>>(pbf, l1bf, pnorm, l1n, sums + 64);

    finalize<<<1, 64, 0, stream>>>(sums, sums + 64, out);
}
